// CausalMultiHeadAttention_85083302133907
// MI455X (gfx1250) — compile-verified, matrix operand order checked by reading, not run on hardware
//
#include <hip/hip_runtime.h>

typedef __attribute__((ext_vector_type(16))) __bf16 v16bf;
typedef __attribute__((ext_vector_type(8)))  float  v8f;
typedef __attribute__((ext_vector_type(4)))  unsigned int u32x4;
typedef __attribute__((ext_vector_type(4)))  int i32x4;
typedef __attribute__((ext_vector_type(8)))  int i32x8;

union Frag16 { v16bf v; uint4 u[2]; };

#if defined(__has_builtin)
#if __has_builtin(__builtin_amdgcn_tensor_load_to_lds) && \
    __has_builtin(__builtin_amdgcn_s_wait_tensorcnt)
#define ATHENA_HAVE_TDM 1
#endif
#endif

__device__ __forceinline__ __bf16 f2bf(float f) {
  unsigned u = __builtin_bit_cast(unsigned, f);
  unsigned r = u + 0x7FFFu + ((u >> 16) & 1u);   // round-to-nearest-even
  unsigned short s = (unsigned short)(r >> 16);
  return __builtin_bit_cast(__bf16, s);
}

#ifdef ATHENA_HAVE_TDM
__device__ __forceinline__ unsigned lds_addr_of(const void* p) {
  // generic -> LDS addrspace -> byte offset (addrspacecast then ptrtoint)
  return (unsigned)(unsigned long long)(const __attribute__((address_space(3))) void*)p;
}

// TDM: DMA a 2-D bf16 tile (rows x cols elements, row stride in elements)
// from global memory into LDS. Descriptor per CDNA5 ISA D# groups 0/1:
// count=1, type=2 ("image"), data_size=1 (2B), no gather/iterate/pad/multicast,
// tensor dims == tile dims (tile fully in bounds).
__device__ __forceinline__ void tdm_load_tile_bf16(const __bf16* gsrc, unsigned lds_bytes,
                                                   unsigned cols, unsigned rows,
                                                   unsigned long long stride_elems) {
  unsigned long long ga = (unsigned long long)gsrc;
  u32x4 g0;
  g0[0] = 1u;                                                   // count=1, user D#
  g0[1] = lds_bytes;                                            // lds_addr
  g0[2] = (unsigned)(ga & 0xFFFFFFFFull);                       // global_addr[31:0]
  g0[3] = (unsigned)((ga >> 32) & 0x01FFFFFFull) | (2u << 30);  // addr[56:32] | type=2
  i32x8 g1;
  g1[0] = (int)(1u << 16);                                      // data_size=1 (2 bytes)
  g1[1] = (int)((cols & 0xFFFFu) << 16);                        // tensor_dim0[15:0]
  g1[2] = (int)(((cols >> 16) & 0xFFFFu) | ((rows & 0xFFFFu) << 16)); // td0 hi | td1 lo
  g1[3] = (int)(((rows >> 16) & 0xFFFFu) | ((cols & 0xFFFFu) << 16)); // td1 hi | tile_dim0
  g1[4] = (int)(rows & 0xFFFFu);                                // tile_dim1 (tile_dim2=0)
  g1[5] = (int)(unsigned)(stride_elems & 0xFFFFFFFFull);        // dim0_stride[31:0]
  g1[6] = (int)(unsigned)((stride_elems >> 32) & 0xFFFFull);    // dim0_stride[47:32]
  g1[7] = 0;                                                    // dim1_stride hi
  i32x4 z4 = {};
#if defined(__clang_major__) && (__clang_major__ >= 23)
  i32x8 z8 = {};
  __builtin_amdgcn_tensor_load_to_lds(g0, g1, z4, z4, z8, 0);
#else
  __builtin_amdgcn_tensor_load_to_lds(g0, g1, z4, z4, 0);
#endif
}
#endif  // ATHENA_HAVE_TDM

// ---------------------------------------------------------------------------
// fp32 -> bf16 downconvert, 4 elements/thread
// ---------------------------------------------------------------------------
__global__ __launch_bounds__(256) void cvt_f32_to_bf16(const float* __restrict__ in,
                                                       __bf16* __restrict__ out, int n4) {
  int i = blockIdx.x * 256 + threadIdx.x;
  if (i < n4) {
    float4 f = ((const float4*)in)[i];
    union { __bf16 h[4]; uint2 u; } o;
    o.h[0] = f2bf(f.x); o.h[1] = f2bf(f.y); o.h[2] = f2bf(f.z); o.h[3] = f2bf(f.w);
    ((uint2*)out)[i] = o.u;
  }
}

// ---------------------------------------------------------------------------
// C = A(MxK bf16 row-major) @ W(NxK bf16 row-major)^T
// 128x128 block tile, 8 wave32 waves (4m x 2n), 32x64 per wave, K-step 32.
// Double-buffered LDS tiles, one barrier per K-step; next tile staged by TDM
// (one wave) or by pipelined cooperative loads. WMMAs overlap the staging.
// mode 0: f32 row-major   mode 1: bf16 row-major   mode 2: bf16 (B,H,S,hd)
// ---------------------------------------------------------------------------
__global__ __launch_bounds__(256) void gemm_xwT(const __bf16* __restrict__ A,
                                                const __bf16* __restrict__ W,
                                                void* __restrict__ Cout,
                                                int M, int N, int K,
                                                int mode, int S, int H, int HD) {
  __shared__ __bf16 As[2][128][32];
  __shared__ __bf16 Bs[2][128][32];
  const int tid  = threadIdx.x;
  const int lane = tid & 31, wid = tid >> 5;
  const int half = lane >> 4, l16 = lane & 15;
  const int wm = wid & 3, wn = wid >> 2;
  const int bm = blockIdx.x * 128, bn = blockIdx.y * 128;
  const int T = K / 32;

  v8f acc[2][4];
  #pragma unroll
  for (int i = 0; i < 2; ++i)
    #pragma unroll
    for (int j = 0; j < 4; ++j) acc[i][j] = {};

  // ---- stage tile 0 into buffer 0 ----
#ifdef ATHENA_HAVE_TDM
  if (wid == 0) {
    tdm_load_tile_bf16(A + (size_t)bm * K, lds_addr_of(&As[0][0][0]), 32u, 128u,
                       (unsigned long long)K);
    tdm_load_tile_bf16(W + (size_t)bn * K, lds_addr_of(&Bs[0][0][0]), 32u, 128u,
                       (unsigned long long)K);
    __builtin_amdgcn_s_wait_tensorcnt(0);
  }
#else
  #pragma unroll
  for (int i = 0; i < 2; ++i) {
    int c = tid + i * 256, row = c >> 2, off = (c & 3) * 8;
    *(uint4*)(&As[0][row][off]) = *(const uint4*)(A + (size_t)(bm + row) * K + off);
    *(uint4*)(&Bs[0][row][off]) = *(const uint4*)(W + (size_t)(bn + row) * K + off);
  }
#endif

  for (int t = 0; t < T; ++t) {
    const int buf = t & 1;
    const int k0 = t * 32;
    const bool hasnext = (t + 1 < T);
    __syncthreads();  // publish buf; all readers of buf^1 are done

    // issue staging for tile t+1 into buf^1 (overlaps with WMMAs below)
#ifdef ATHENA_HAVE_TDM
    if (hasnext && wid == 0) {
      tdm_load_tile_bf16(A + (size_t)bm * K + (k0 + 32), lds_addr_of(&As[buf ^ 1][0][0]),
                         32u, 128u, (unsigned long long)K);
      tdm_load_tile_bf16(W + (size_t)bn * K + (k0 + 32), lds_addr_of(&Bs[buf ^ 1][0][0]),
                         32u, 128u, (unsigned long long)K);
    }
#else
    uint4 ra[2], rb[2];
    if (hasnext) {
      #pragma unroll
      for (int i = 0; i < 2; ++i) {
        int c = tid + i * 256, row = c >> 2, off = (c & 3) * 8;
        ra[i] = *(const uint4*)(A + (size_t)(bm + row) * K + k0 + 32 + off);
        rb[i] = *(const uint4*)(W + (size_t)(bn + row) * K + k0 + 32 + off);
      }
    }
#endif

    // ---- compute from buf ----
    Frag16 af[2], bfr[4];
    #pragma unroll
    for (int i = 0; i < 2; ++i) {
      int r = wm * 32 + i * 16 + l16;
      af[i].u[0] = *(const uint4*)(&As[buf][r][half * 8]);
      af[i].u[1] = *(const uint4*)(&As[buf][r][16 + half * 8]);
    }
    #pragma unroll
    for (int j = 0; j < 4; ++j) {
      int r = wn * 64 + j * 16 + l16;
      bfr[j].u[0] = *(const uint4*)(&Bs[buf][r][half * 8]);
      bfr[j].u[1] = *(const uint4*)(&Bs[buf][r][16 + half * 8]);
    }
    #pragma unroll
    for (int i = 0; i < 2; ++i)
      #pragma unroll
      for (int j = 0; j < 4; ++j)
        acc[i][j] = __builtin_amdgcn_wmma_f32_16x16x32_bf16(
            false, af[i].v, false, bfr[j].v, (short)0, acc[i][j], false, false);

    // ---- finish staging tile t+1 ----
#ifdef ATHENA_HAVE_TDM
    if (hasnext && wid == 0) __builtin_amdgcn_s_wait_tensorcnt(0);
#else
    if (hasnext) {
      #pragma unroll
      for (int i = 0; i < 2; ++i) {
        int c = tid + i * 256, row = c >> 2, off = (c & 3) * 8;
        *(uint4*)(&As[buf ^ 1][row][off]) = ra[i];
        *(uint4*)(&Bs[buf ^ 1][row][off]) = rb[i];
      }
    }
#endif
  }

  #pragma unroll
  for (int i = 0; i < 2; ++i)
    #pragma unroll
    for (int j = 0; j < 4; ++j)
      #pragma unroll
      for (int r = 0; r < 8; ++r) {
        int row = bm + wm * 32 + i * 16 + r + 8 * half;   // D-layout: VGPR r, lane half
        int col = bn + wn * 64 + j * 16 + l16;
        float v = acc[i][j][r];
        if (mode == 0) {
          ((float*)Cout)[(size_t)row * N + col] = v;
        } else if (mode == 1) {
          ((__bf16*)Cout)[(size_t)row * N + col] = f2bf(v);
        } else {
          int b = row / S, s = row - b * S;
          int h = col / HD, d = col - h * HD;
          ((__bf16*)Cout)[(((size_t)(b * H + h)) * S + s) * HD + d] = f2bf(v);
        }
      }
}

// ---------------------------------------------------------------------------
// Fused attention on (B,H,S,64) bf16 Q/K/V; reference's mask bug preserved:
// score kept only where key > query (strictly), else -1e9.
// Block = 128 query rows (8 waves x 16). 32-key tiles, online softmax.
// Double-buffered K/V LDS tiles: K staged by TDM (or pipelined coop loads),
// V staged transposed via pipelined cooperative loads.
// ---------------------------------------------------------------------------
__global__ __launch_bounds__(256) void attn_fused(const __bf16* __restrict__ Q,
                                                  const __bf16* __restrict__ Km,
                                                  const __bf16* __restrict__ Vm,
                                                  __bf16* __restrict__ Out,
                                                  int H, int S) {
  __shared__ __bf16 Ks[2][32][64];    // keys x head_dim
  __shared__ __bf16 VsT[2][64][32];   // head_dim x keys (transposed)
  __shared__ __bf16 Ps[8][16][32];    // per-wave P scratch (D->A relayout)
  const int tid  = threadIdx.x;
  const int lane = tid & 31, wid = tid >> 5;
  const int half = lane >> 4, l16 = lane & 15;
  const int bh = blockIdx.y;
  const int q0 = blockIdx.x * 128;
  const int HD = 64;
  const size_t headbase = (size_t)bh * S * HD;
  const int vrow = tid >> 3;          // 0..31
  const int voff = (tid & 7) * 8;     // 0..56

  // Q fragments (A operand), 16 rows x 64 k, two K=32 chunks
  Frag16 qa[2];
  {
    const __bf16* qp = Q + headbase + (size_t)(q0 + wid * 16 + l16) * HD;
    #pragma unroll
    for (int c = 0; c < 2; ++c) {
      qa[c].u[0] = *(const uint4*)(qp + c * 32 + half * 8);
      qa[c].u[1] = *(const uint4*)(qp + c * 32 + 16 + half * 8);
    }
  }

  float mrow[8], lrow[8];
  v8f acc[4];
  #pragma unroll
  for (int r = 0; r < 8; ++r) { mrow[r] = -3.0e38f; lrow[r] = 0.f; }
  #pragma unroll
  for (int j = 0; j < 4; ++j) acc[j] = {};

  // ---- stage tile 0 ----
#ifdef ATHENA_HAVE_TDM
  if (wid == 0)
    tdm_load_tile_bf16(Km + headbase, lds_addr_of(&Ks[0][0][0]), 64u, 32u, 64ull);
#else
  *(uint4*)(&Ks[0][vrow][voff]) =
      *(const uint4*)(Km + headbase + (size_t)vrow * HD + voff);
#endif
  {
    union { uint4 u; __bf16 h[8]; } tv;
    tv.u = *(const uint4*)(Vm + headbase + (size_t)vrow * HD + voff);
    #pragma unroll
    for (int e = 0; e < 8; ++e) VsT[0][voff + e][vrow] = tv.h[e];
  }
#ifdef ATHENA_HAVE_TDM
  if (wid == 0) __builtin_amdgcn_s_wait_tensorcnt(0);
#endif

  const float scale = 0.03125f;  // 1/sqrt(1024) — module scales by sqrt(d_model)
  const int T = S / 32;

  for (int t = 0; t < T; ++t) {
    const int buf = t & 1;
    const int kt0 = t * 32;
    const bool hasnext = (t + 1 < T);
    __syncthreads();  // publish buf

    // issue next-tile staging (overlaps the compute below)
    uint4 vnext;
#ifdef ATHENA_HAVE_TDM
    if (hasnext && wid == 0)
      tdm_load_tile_bf16(Km + headbase + (size_t)(kt0 + 32) * HD,
                         lds_addr_of(&Ks[buf ^ 1][0][0]), 64u, 32u, 64ull);
#else
    uint4 knext;
    if (hasnext)
      knext = *(const uint4*)(Km + headbase + (size_t)(kt0 + 32 + vrow) * HD + voff);
#endif
    if (hasnext)
      vnext = *(const uint4*)(Vm + headbase + (size_t)(kt0 + 32 + vrow) * HD + voff);

    // ---- scores: two 16x16 key tiles, K=64 via two chained WMMAs each ----
    Frag16 kf[2][2];
    v8f sc[2];
    #pragma unroll
    for (int g = 0; g < 2; ++g) {
      #pragma unroll
      for (int c = 0; c < 2; ++c) {
        int kr = g * 16 + l16;
        kf[g][c].u[0] = *(const uint4*)(&Ks[buf][kr][c * 32 + half * 8]);
        kf[g][c].u[1] = *(const uint4*)(&Ks[buf][kr][c * 32 + 16 + half * 8]);
      }
      sc[g] = {};
      sc[g] = __builtin_amdgcn_wmma_f32_16x16x32_bf16(
          false, qa[0].v, false, kf[g][0].v, (short)0, sc[g], false, false);
      sc[g] = __builtin_amdgcn_wmma_f32_16x16x32_bf16(
          false, qa[1].v, false, kf[g][1].v, (short)0, sc[g], false, false);
    }

    // scale + (buggy) mask: keep strictly-future keys, else -1e9
    #pragma unroll
    for (int g = 0; g < 2; ++g)
      #pragma unroll
      for (int r = 0; r < 8; ++r) {
        int col = kt0 + g * 16 + l16;
        int qg  = q0 + wid * 16 + r + 8 * half;
        float s = sc[g][r] * scale;
        sc[g][r] = (col > qg) ? s : -1.0e9f;
      }

    // online softmax; shfl_xor 1/2/4/8 stays within the row-owning 16-lane half
    float sf[8];
    #pragma unroll
    for (int r = 0; r < 8; ++r) {
      float mx = fmaxf(sc[0][r], sc[1][r]);
      #pragma unroll
      for (int xm = 1; xm <= 8; xm <<= 1) mx = fmaxf(mx, __shfl_xor(mx, xm, 32));
      float mn = fmaxf(mrow[r], mx);
      sf[r] = __expf(mrow[r] - mn);
      mrow[r] = mn;
      float p0 = __expf(sc[0][r] - mn);
      float p1 = __expf(sc[1][r] - mn);
      sc[0][r] = p0; sc[1][r] = p1;
      float rs = p0 + p1;
      #pragma unroll
      for (int xm = 1; xm <= 8; xm <<= 1) rs += __shfl_xor(rs, xm, 32);
      lrow[r] = lrow[r] * sf[r] + rs;
    }
    #pragma unroll
    for (int j = 0; j < 4; ++j)
      #pragma unroll
      for (int r = 0; r < 8; ++r) acc[j][r] *= sf[r];

    // P: D-layout -> A-layout via per-wave LDS scratch (in-order DS per wave)
    #pragma unroll
    for (int g = 0; g < 2; ++g)
      #pragma unroll
      for (int r = 0; r < 8; ++r)
        Ps[wid][r + 8 * half][g * 16 + l16] = f2bf(sc[g][r]);
    Frag16 pa;
    pa.u[0] = *(const uint4*)(&Ps[wid][l16][half * 8]);
    pa.u[1] = *(const uint4*)(&Ps[wid][l16][16 + half * 8]);

    // O += P @ V  (4 output 16-col groups)
    #pragma unroll
    for (int j = 0; j < 4; ++j) {
      Frag16 vf;
      int n = j * 16 + l16;
      vf.u[0] = *(const uint4*)(&VsT[buf][n][half * 8]);
      vf.u[1] = *(const uint4*)(&VsT[buf][n][16 + half * 8]);
      acc[j] = __builtin_amdgcn_wmma_f32_16x16x32_bf16(
          false, pa.v, false, vf.v, (short)0, acc[j], false, false);
    }

    // ---- finish staging tile t+1 into buf^1 ----
    if (hasnext) {
      union { uint4 u; __bf16 h[8]; } tv;
      tv.u = vnext;
      #pragma unroll
      for (int e = 0; e < 8; ++e) VsT[buf ^ 1][voff + e][vrow] = tv.h[e];
#ifndef ATHENA_HAVE_TDM
      *(uint4*)(&Ks[buf ^ 1][vrow][voff]) = knext;
#endif
    }
#ifdef ATHENA_HAVE_TDM
    if (hasnext && wid == 0) __builtin_amdgcn_s_wait_tensorcnt(0);
#endif
  }

  // epilogue: normalize, write (B,S,D) bf16 with D = h*64 + col
  int b = bh / H, h = bh - b * H;
  #pragma unroll
  for (int j = 0; j < 4; ++j)
    #pragma unroll
    for (int r = 0; r < 8; ++r) {
      int row = q0 + wid * 16 + r + 8 * half;
      int col = j * 16 + l16;
      float v = acc[j][r] / lrow[r];
      Out[((size_t)(b * S + row)) * (H * HD) + h * HD + col] = f2bf(v);
    }
}

// ---------------------------------------------------------------------------
extern "C" void kernel_launch(void* const* d_in, const int* in_sizes, int n_in,
                              void* d_out, int out_size, void* d_ws, size_t ws_size,
                              hipStream_t stream) {
  const float* x  = (const float*)d_in[0];
  const float* wq = (const float*)d_in[1];
  const float* wk = (const float*)d_in[2];
  const float* wv = (const float*)d_in[3];
  const float* wo = (const float*)d_in[4];
  (void)in_sizes; (void)n_in; (void)out_size; (void)ws_size;

  const int B = 4, S = 2048, D = 1024, H = 16, HD = 64;
  const int M = B * S;  // 8192

  char* ws = (char*)d_ws;
  size_t off = 0;
  auto carve = [&](size_t bytes) {
    char* p = ws + off;
    off += (bytes + 255) & ~(size_t)255;
    return p;
  };
  __bf16* xb  = (__bf16*)carve((size_t)M * D * 2);   // 16 MB
  __bf16* wqb = (__bf16*)carve((size_t)D * D * 2);   //  2 MB
  __bf16* wkb = (__bf16*)carve((size_t)D * D * 2);
  __bf16* wvb = (__bf16*)carve((size_t)D * D * 2);
  __bf16* wob = (__bf16*)carve((size_t)D * D * 2);
  __bf16* qb  = (__bf16*)carve((size_t)M * D * 2);   // (B,H,S,hd)
  __bf16* kb  = (__bf16*)carve((size_t)M * D * 2);
  __bf16* vb  = (__bf16*)carve((size_t)M * D * 2);
  __bf16* ab  = (__bf16*)carve((size_t)M * D * 2);   // attn out (B,S,D)

  {
    int n4 = M * D / 4;
    cvt_f32_to_bf16<<<(n4 + 255) / 256, 256, 0, stream>>>(x, xb, n4);
    int w4 = D * D / 4;
    cvt_f32_to_bf16<<<(w4 + 255) / 256, 256, 0, stream>>>(wq, wqb, w4);
    cvt_f32_to_bf16<<<(w4 + 255) / 256, 256, 0, stream>>>(wk, wkb, w4);
    cvt_f32_to_bf16<<<(w4 + 255) / 256, 256, 0, stream>>>(wv, wvb, w4);
    cvt_f32_to_bf16<<<(w4 + 255) / 256, 256, 0, stream>>>(wo, wob, w4);
  }

  dim3 gg(M / 128, D / 128);
  gemm_xwT<<<gg, 256, 0, stream>>>(xb, wqb, qb, M, D, D, 2, S, H, HD);
  gemm_xwT<<<gg, 256, 0, stream>>>(xb, wkb, kb, M, D, D, 2, S, H, HD);
  gemm_xwT<<<gg, 256, 0, stream>>>(xb, wvb, vb, M, D, D, 2, S, H, HD);

  attn_fused<<<dim3(S / 128, B * H), 256, 0, stream>>>(qb, kb, vb, ab, H, S);

  gemm_xwT<<<gg, 256, 0, stream>>>(ab, wob, d_out, M, D, D, 0, S, H, HD);
}
